// Mymodel_37847251812630
// MI455X (gfx1250) — matrix operand order repaired and checked by reading, not compiled, not run on hardware
//
#include <hip/hip_runtime.h>
#include <hip/hip_bf16.h>

typedef __attribute__((ext_vector_type(16))) __bf16 v16bf;
typedef __attribute__((ext_vector_type(8)))  __bf16 v8bf;
typedef __attribute__((ext_vector_type(8)))  float  v8f;

// ---------------------------------------------------------------------------
// fp32 -> bf16 (RNE) weight conversion into padded [Cout16][K32] layout,
// zero-filled so the GEMM needs no bounds checks on the A (weight) stream.
// ---------------------------------------------------------------------------
__global__ void cvt_w_bf16(const float* __restrict__ src,
                           unsigned short* __restrict__ dst,
                           int Cout, int K, int Kpad, int n) {
  int i = blockIdx.x * blockDim.x + threadIdx.x;
  if (i >= n) return;
  int cout = i / Kpad;
  int k    = i - cout * Kpad;
  unsigned short r = 0;
  if (cout < Cout && k < K) {
    unsigned u = __builtin_bit_cast(unsigned, src[cout * K + k]);
    r = (unsigned short)((u + 0x7FFFu + ((u >> 16) & 1u)) >> 16);
  }
  dst[i] = r;
}

// ---------------------------------------------------------------------------
// Implicit-GEMM conv via V_WMMA_F32_16X16X32_BF16, 4x N-blocking,
// double-buffered LDS staging (1 barrier per K-step, loads overlap WMMAs).
// GEMM view: M = Cout, N = B*H*W, K = Cin*KS*KS (padded to 32).
// Block = 8 waves = 8 cout-tiles, all sharing 4 pixel-tiles (128x64 output).
// Staged element address = per-pixel base (LDS) + per-k offset (register).
// ---------------------------------------------------------------------------
template <int KS>
__global__ __launch_bounds__(256) void conv_wmma(
    const float* __restrict__ in, const unsigned short* __restrict__ wgt,
    const float* __restrict__ bias, float* __restrict__ out,
    int B, int Cin, int Cout, int H, int W, int pad, int dil, int relu)
{
  const int K    = Cin * KS * KS;
  const int Kpad = (K + 31) & ~31;
  const int HW   = H * W;
  const int Ntot = B * HW;
  const int Mt   = (Cout + 15) >> 4;
  const int Nt   = (Ntot + 15) >> 4;
  const int nBlocks = (Nt + 3) >> 2;

  const int bid = (int)blockIdx.x;
  const int bm  = bid / nBlocks;
  const int bn  = bid - bm * nBlocks;

  const int tid   = (int)threadIdx.x;
  const int wave  = tid >> 5;        // cout tile within block (0..7)
  const int lane  = tid & 31;
  const int lhalf = lane >> 4;
  const int l15   = lane & 15;

  const int mt  = bm * 8 + wave;
  const int mtc = (mt < Mt) ? mt : (Mt - 1);

  __shared__ __attribute__((aligned(32))) __bf16 sB[2][4][16][32];
  __shared__ int s_pb[64], s_ph[64], s_pw[64], s_pbase[64];

  // Per-block pixel coords + flat input base (64 pixels = 4 N-tiles), once.
  if (tid < 64) {
    int nIdx = bn * 64 + tid;
    if (nIdx >= Ntot) nIdx = 0;              // clamp (store is guarded)
    int b  = nIdx / HW;
    int r  = nIdx - b * HW;
    int oh = r / W;
    int ow = r - oh * W;
    s_pb[tid]    = b;
    s_ph[tid]    = oh;
    s_pw[tid]    = ow;
    s_pbase[tid] = b * Cin * HW + oh * W + ow;   // < 2^31 for all layers
  }

  // Stage one 64px x 32k bf16 tile; each thread owns k = lane for 8 pixels.
  auto stage = [&](int k0s, int buf) {
    int kk = k0s + lane;
    int kc = (kk < K) ? kk : (K - 1);
    int cin = kc / (KS * KS);
    int r   = kc - cin * (KS * KS);
    int kh  = r / KS;
    int kw  = r - kh * KS;
    int dh  = kh * dil - pad;
    int dw  = kw * dil - pad;
    int koff = cin * HW + dh * W + dw;       // per-k flat offset
    bool kOK = (kk < K);
    #pragma unroll
    for (int e = 0; e < 8; ++e) {
      int pi = wave + 8 * e;                 // 0..63, exact cover
      float v = 0.0f;
      int ih = s_ph[pi] + dh;
      int iw = s_pw[pi] + dw;
      if (kOK && (unsigned)ih < (unsigned)H && (unsigned)iw < (unsigned)W)
        v = in[s_pbase[pi] + koff];
      sB[buf][pi >> 4][pi & 15][lane] = (__bf16)v;
    }
  };

  // A stream base for this lane: row = mtc*16 + l15, k = 16*lhalf + 0..15 (contiguous).
  const size_t wbase = (size_t)(mtc * 16 + l15) * (size_t)Kpad + (size_t)(lhalf * 16);

  __syncthreads();                           // pixel coords visible
  stage(0, 0);                               // prologue
  __syncthreads();

  v8f acc[4] = {};
  int cb = 0;
  for (int k0 = 0; k0 < Kpad; k0 += 32) {
    // Stage next K-step into the ping buffer (overlaps with WMMAs below).
    if (k0 + 32 < Kpad) stage(k0 + 32, cb ^ 1);

    // A fragment: two aligned b128 loads (no guards, padded buffer).
    const unsigned short* pw = wgt + wbase + k0;
    v8bf lo = *(const v8bf*)pw;
    v8bf hi = *(const v8bf*)(pw + 8);
    __builtin_prefetch(pw + 32, 0, 1);       // next K-step of the weight row
    v16bf a;
    #pragma unroll
    for (int i = 0; i < 8; ++i) { a[i] = lo[i]; a[8 + i] = hi[i]; }

    // 4 WMMAs reusing A against the 4 staged pixel tiles (pong buffer).
    #pragma unroll
    for (int j = 0; j < 4; ++j) {
      v16bf bf = *(const v16bf*)&sB[cb][j][l15][lhalf * 16];
      acc[j] = __builtin_amdgcn_wmma_f32_16x16x32_bf16(
          false, a, false, bf, (short)0, acc[j], false, false);
    }

    __syncthreads();                         // single barrier per K-step
    cb ^= 1;
  }

  // ---- Store (guarded; coords from LDS; bias hoisted) ---------------------
  if (mt < Mt) {
    float bv[8];
    #pragma unroll
    for (int r = 0; r < 8; ++r) {
      int cout = mt * 16 + r + lhalf * 8;
      bv[r] = (cout < Cout) ? bias[cout] : 0.0f;
    }
    #pragma unroll
    for (int j = 0; j < 4; ++j) {
      int nIdx = (bn * 4 + j) * 16 + l15;
      if (nIdx < Ntot) {
        const int pi = j * 16 + l15;
        const int bImg = s_pb[pi], oh = s_ph[pi], ow = s_pw[pi];
        #pragma unroll
        for (int r = 0; r < 8; ++r) {
          int cout = mt * 16 + r + lhalf * 8;  // C/D layout: M = r + 8*lane_half
          if (cout < Cout) {
            float v = acc[j][r] + bv[r];
            if (relu) v = fmaxf(v, 0.0f);
            out[(((size_t)bImg * Cout + cout) * H + oh) * W + ow] = v;
          }
        }
      }
    }
  }
}

// ---------------------------------------------------------------------------
// Generic max-pool (floor/ceil via host-computed Hout; OOB skip == -inf pad).
// ---------------------------------------------------------------------------
__global__ void maxpool_k(const float* __restrict__ in, float* __restrict__ out,
                          int B, int C, int Hin, int Win, int Hout, int Wout,
                          int k, int s, int pad) {
  int idx = blockIdx.x * blockDim.x + threadIdx.x;
  int tot = B * C * Hout * Wout;
  if (idx >= tot) return;
  int ow = idx % Wout; int t = idx / Wout;
  int oh = t % Hout;   t /= Hout;
  int c  = t % C;      int b = t / C;
  const float* p = in + (size_t)(b * C + c) * Hin * Win;
  float m = -INFINITY;
  int h0 = oh * s - pad, w0 = ow * s - pad;
  for (int i = 0; i < k; ++i) {
    int ih = h0 + i;
    if ((unsigned)ih >= (unsigned)Hin) continue;
    for (int j = 0; j < k; ++j) {
      int iw = w0 + j;
      if ((unsigned)iw >= (unsigned)Win) continue;
      m = fmaxf(m, p[ih * Win + iw]);
    }
  }
  out[idx] = m;
}

// ---------------------------------------------------------------------------
// Head reorder: [B,C,19,19] -> SSD flattened layout, plus priors copy.
// ---------------------------------------------------------------------------
__global__ void write_loc_k(const float* __restrict__ loc, float* __restrict__ out) {
  int i = blockIdx.x * blockDim.x + threadIdx.x;     // over B*361*16
  const int tot = 4 * 361 * 16;
  if (i >= tot) return;
  int c = i % 16; int t = i / 16;
  int p = t % 361; int b = t / 361;
  int a = c >> 2, coord = c & 3;
  out[((b * 1444 + p * 4 + a) * 4) + coord] = loc[((size_t)(b * 16 + c)) * 361 + p];
}

__global__ void write_conf_k(const float* __restrict__ conf, float* __restrict__ out) {
  int i = blockIdx.x * blockDim.x + threadIdx.x;     // over B*361*84
  const int tot = 4 * 361 * 84;
  if (i >= tot) return;
  int c = i % 84; int t = i / 84;
  int p = t % 361; int b = t / 361;
  int a = c / 21, cls = c - a * 21;
  out[23104 + ((b * 1444 + p * 4 + a) * 21) + cls] = conf[((size_t)(b * 84 + c)) * 361 + p];
}

__global__ void write_priors_k(const float* __restrict__ pri, float* __restrict__ out) {
  int i = blockIdx.x * blockDim.x + threadIdx.x;
  if (i < 5776) out[144400 + i] = pri[i];
}

// ---------------------------------------------------------------------------
// Host orchestration
// ---------------------------------------------------------------------------
struct ConvCfg { int widx, cin, cout, hw, ks, pad, dil, relu; };

extern "C" void kernel_launch(void* const* d_in, const int* in_sizes, int n_in,
                              void* d_out, int out_size, void* d_ws, size_t ws_size,
                              hipStream_t stream) {
  (void)in_sizes; (void)n_in; (void)out_size; (void)ws_size;

  static const ConvCfg cv[17] = {
    { 1,    3,   64, 300, 3, 1, 1, 1},
    { 3,   64,   64, 300, 3, 1, 1, 1},
    { 5,   64,  128, 150, 3, 1, 1, 1},
    { 7,  128,  128, 150, 3, 1, 1, 1},
    { 9,  128,  256,  75, 3, 1, 1, 1},
    {11,  256,  256,  75, 3, 1, 1, 1},
    {13,  256,  256,  75, 3, 1, 1, 1},
    {15,  256,  512,  38, 3, 1, 1, 1},
    {17,  512,  512,  38, 3, 1, 1, 1},
    {19,  512,  512,  38, 3, 1, 1, 1},
    {21,  512,  512,  19, 3, 1, 1, 1},
    {23,  512,  512,  19, 3, 1, 1, 1},
    {25,  512,  512,  19, 3, 1, 1, 1},
    {27,  512, 1024,  19, 3, 6, 6, 1},   // conv6 (dilated)
    {29, 1024, 1024,  19, 1, 0, 1, 1},   // conv7 (1x1)
    {31, 1024,   16,  19, 3, 1, 1, 0},   // loc head
    {33, 1024,   84,  19, 3, 1, 1, 0},   // conf head
  };

  const float* x = (const float*)d_in[0];
  float* out = (float*)d_out;

  // Workspace: two fp32 activation ping-pong buffers + padded bf16 weights.
  const size_t ACT = 23040000ull;           // 4*64*300*300 floats (largest act)
  float* bufA = (float*)d_ws;
  float* bufB = bufA + ACT;
  unsigned short* wb = (unsigned short*)(bufB + ACT);

  size_t woff[17]; size_t o = 0;
  int kpadv[17], cpadv[17];
  for (int i = 0; i < 17; ++i) {
    int K    = cv[i].cin * cv[i].ks * cv[i].ks;
    int Kpad = (K + 31) & ~31;
    int Cpad = (cv[i].cout + 15) & ~15;
    kpadv[i] = Kpad; cpadv[i] = Cpad;
    woff[i] = o;
    o += (size_t)Cpad * Kpad;
  }
  for (int i = 0; i < 17; ++i) {
    int K = cv[i].cin * cv[i].ks * cv[i].ks;
    int n = cpadv[i] * kpadv[i];
    cvt_w_bf16<<<(n + 255) / 256, 256, 0, stream>>>(
        (const float*)d_in[cv[i].widx], wb + woff[i],
        cv[i].cout, K, kpadv[i], n);
  }

  auto runconv = [&](int idx, const float* inp, float* outp) {
    const ConvCfg& c = cv[idx];
    int Ntot = 4 * c.hw * c.hw;
    int Mt = (c.cout + 15) / 16;
    int Nt = (Ntot + 15) / 16;
    int blocks = ((Mt + 7) / 8) * ((Nt + 3) / 4);
    if (c.ks == 3) {
      conv_wmma<3><<<blocks, 256, 0, stream>>>(
          inp, wb + woff[idx], (const float*)d_in[c.widx + 1], outp,
          4, c.cin, c.cout, c.hw, c.hw, c.pad, c.dil, c.relu);
    } else {
      conv_wmma<1><<<blocks, 256, 0, stream>>>(
          inp, wb + woff[idx], (const float*)d_in[c.widx + 1], outp,
          4, c.cin, c.cout, c.hw, c.hw, c.pad, c.dil, c.relu);
    }
  };
  auto runpool = [&](const float* inp, float* outp, int C, int Hin, int Hout,
                     int k, int s, int pad) {
    int tot = 4 * C * Hout * Hout;
    maxpool_k<<<(tot + 255) / 256, 256, 0, stream>>>(
        inp, outp, 4, C, Hin, Hin, Hout, Hout, k, s, pad);
  };

  // VGG backbone
  runconv(0, x,    bufA);                        // 3->64   @300
  runconv(1, bufA, bufB);                        // 64->64  @300
  runpool(bufB, bufA, 64, 300, 150, 2, 2, 0);    // M
  runconv(2, bufA, bufB);                        // 64->128 @150
  runconv(3, bufB, bufA);                        // 128->128
  runpool(bufA, bufB, 128, 150, 75, 2, 2, 0);    // M
  runconv(4, bufB, bufA);                        // 128->256 @75
  runconv(5, bufA, bufB);
  runconv(6, bufB, bufA);
  runpool(bufA, bufB, 256, 75, 38, 2, 2, 0);     // 'C' ceil pool -> 38
  runconv(7, bufB, bufA);                        // 256->512 @38
  runconv(8, bufA, bufB);
  runconv(9, bufB, bufA);
  runpool(bufA, bufB, 512, 38, 19, 2, 2, 0);     // M
  runconv(10, bufB, bufA);                       // 512->512 @19
  runconv(11, bufA, bufB);
  runconv(12, bufB, bufA);
  runpool(bufA, bufB, 512, 19, 19, 3, 1, 1);     // pool5 (k3,s1,p1)
  runconv(13, bufB, bufA);                       // conv6 dilated -> 1024 @19
  runconv(14, bufA, bufB);                       // conv7 1x1 -> 1024 @19

  // Heads: loc -> bufA[0 .. 23104), conf -> bufA + 23104
  float* locbuf  = bufA;
  float* confbuf = bufA + 4 * 16 * 361;
  runconv(15, bufB, locbuf);
  runconv(16, bufB, confbuf);

  // Assemble output: loc | conf | priors
  {
    int tot = 4 * 361 * 16;
    write_loc_k<<<(tot + 255) / 256, 256, 0, stream>>>(locbuf, out);
  }
  {
    int tot = 4 * 361 * 84;
    write_conf_k<<<(tot + 255) / 256, 256, 0, stream>>>(confbuf, out);
  }
  write_priors_k<<<(5776 + 255) / 256, 256, 0, stream>>>(
      (const float*)d_in[35], out);
}
